// CrossViewAttention_21947282883267
// MI455X (gfx1250) — matrix operand order repaired and checked by reading, not compiled, not run on hardware
//
#include <hip/hip_runtime.h>
#include <hip/hip_bf16.h>
#include <math.h>

// CrossViewAttention for MI455X (gfx1250, wave32).
// All GEMM-shaped work lowered to v_wmma_f32_16x16x32_bf16 (f32->bf16 inputs,
// f32 accumulation). Flash-attention over the joint n*K softmax axis.

typedef __attribute__((ext_vector_type(16))) __bf16 v16bf;
typedef __attribute__((ext_vector_type(8)))  float  v8f;

#define DEV __device__ __forceinline__

constexpr int cB  = 2;
constexpr int cN  = 6;
constexpr int cBN = 12;
constexpr int cCF = 128;
constexpr int cK  = 1680;   // 28*60 keys per camera
constexpr int cD  = 128;
constexpr int cMH = 4;
constexpr int cDH = 32;
constexpr int cQ  = 1024;   // 32*32 BEV queries
constexpr float cEPS = 1e-5f;

DEV v8f v8f_zero() {
  v8f z = {0.f, 0.f, 0.f, 0.f, 0.f, 0.f, 0.f, 0.f};
  return z;
}

DEV v8f wmma_bf16(v16bf a, v16bf b, v8f c) {
  // D = A(16x32) * B(32x16) + C, f32 accumulate
  return __builtin_amdgcn_wmma_f32_16x16x32_bf16(false, a, false, b, (short)0, c,
                                                 false, false);
}

// A fragment (16x32 bf16) gathered from LDS, row stride ld, K offset k0.
// Layout: lane L<16 -> row L, K in {0..15}; lane L+16 -> K in {16..31}.
DEV v16bf lds_a_frag(const __bf16* base, int ld, int k0) {
  int lane = threadIdx.x & 31;
  int row = lane & 15;
  int ko = (lane >> 4) << 4;
  v16bf a;
#pragma unroll
  for (int e = 0; e < 16; ++e)
    a[e] = base[row * ld + k0 + e + ko];
  return a;
}

// B fragment (32x16): B[k][n] = src[(k0+k)*ld + n0+n]   (k-major memory)
DEV v16bf glb_b_frag_k(const float* src, int ld, int k0, int n0) {
  int lane = threadIdx.x & 31;
  int col = lane & 15;
  int ko = (lane >> 4) << 4;
  v16bf b;
#pragma unroll
  for (int e = 0; e < 16; ++e)
    b[e] = (__bf16)src[(size_t)(k0 + ko + e) * ld + n0 + col];
  return b;
}

// B fragment (32x16): B[k][n] = src[(n0+n)*ld + k0+k]   (n-major memory, contiguous k)
DEV v16bf glb_b_frag_n(const float* src, int ld, int k0, int n0) {
  int lane = threadIdx.x & 31;
  int col = lane & 15;
  int ko = (lane >> 4) << 4;
  const float* p = src + (size_t)(n0 + col) * ld + k0 + ko;
  v16bf b;
#pragma unroll
  for (int e = 0; e < 16; ++e) b[e] = (__bf16)p[e];
  return b;
}

DEV float red_add16(float v) {
#pragma unroll
  for (int off = 1; off < 16; off <<= 1) v += __shfl_xor(v, off, 16);
  return v;
}
DEV float red_max16(float v) {
#pragma unroll
  for (int off = 1; off < 16; off <<= 1) v = fmaxf(v, __shfl_xor(v, off, 16));
  return v;
}
DEV float red_add32(float v) {
#pragma unroll
  for (int off = 1; off < 32; off <<= 1) v += __shfl_xor(v, off, 32);
  return v;
}

// ---------------------------------------------------------------- geometry ---
__global__ void k_cam_embed(const float* __restrict__ camw,
                            const float* __restrict__ Einv,
                            float* __restrict__ cem) {
  int bn = blockIdx.x, o = threadIdx.x;
  const float* e = Einv + bn * 16;
  float s = 0.f;
#pragma unroll
  for (int i = 0; i < 4; ++i) s += camw[o * 4 + i] * e[i * 4 + 3];
  cem[bn * cD + o] = s;
}

__global__ void k_bev_query(const float* __restrict__ bevw,
                            const float* __restrict__ bevb,
                            const float* __restrict__ grid,
                            const float* __restrict__ cem,
                            const float* __restrict__ x,
                            float* __restrict__ query) {
  __shared__ float red[128];
  int bn = blockIdx.x / cQ, q = blockIdx.x % cQ, o = threadIdx.x;
  float e = bevw[o * 2] * grid[q] + bevw[o * 2 + 1] * grid[cQ + q] + bevb[o] -
            cem[bn * cD + o];
  red[o] = e * e;
  __syncthreads();
  for (int st = 64; st > 0; st >>= 1) {
    if (o < st) red[o] += red[o + st];
    __syncthreads();
  }
  float inv = 1.f / (sqrtf(red[0]) + 1e-7f);
  int b = bn / cN;
  query[(size_t)(bn * cQ + q) * cD + o] = e * inv + x[(size_t)(b * cD + o) * cQ + q];
}

__global__ void k_img_embed(const float* __restrict__ imgw,
                            const float* __restrict__ Iinv,
                            const float* __restrict__ Einv,
                            const float* __restrict__ plane,
                            const float* __restrict__ cem,
                            float* __restrict__ keyb) {
  __shared__ float red[128];
  int bn = blockIdx.x / cK, k = blockIdx.x % cK, o = threadIdx.x;
  const float* Ii = Iinv + bn * 9;
  const float* Ee = Einv + bn * 16;
  float px = plane[k], py = plane[cK + k], pz = plane[2 * cK + k];
  float c0 = Ii[0] * px + Ii[1] * py + Ii[2] * pz;
  float c1 = Ii[3] * px + Ii[4] * py + Ii[5] * pz;
  float c2 = Ii[6] * px + Ii[7] * py + Ii[8] * pz;
  float d0 = Ee[0] * c0 + Ee[1] * c1 + Ee[2] * c2 + Ee[3];
  float d1 = Ee[4] * c0 + Ee[5] * c1 + Ee[6] * c2 + Ee[7];
  float d2 = Ee[8] * c0 + Ee[9] * c1 + Ee[10] * c2 + Ee[11];
  float d3 = Ee[12] * c0 + Ee[13] * c1 + Ee[14] * c2 + Ee[15];
  float e = imgw[o * 4] * d0 + imgw[o * 4 + 1] * d1 + imgw[o * 4 + 2] * d2 +
            imgw[o * 4 + 3] * d3 - cem[bn * cD + o];
  red[o] = e * e;
  __syncthreads();
  for (int st = 64; st > 0; st >>= 1) {
    if (o < st) red[o] += red[o + st];
    __syncthreads();
  }
  float inv = 1.f / (sqrtf(red[0]) + 1e-7f);
  keyb[(size_t)(bn * cK + k) * cD + o] = e * inv;
}

// --------------------------------------------- BN -> ReLU -> 1x1 conv (WMMA) --
// out[(bn,k,o)] (+)= sum_c relu(bn(feature[bn,c,k])) * W[o,c]
__global__ void k_conv_wmma(const float* __restrict__ feat,
                            const float* __restrict__ bng, const float* __restrict__ bnb,
                            const float* __restrict__ bnm, const float* __restrict__ bnv,
                            const float* __restrict__ Wt, float* __restrict__ outp,
                            int accum) {
  __shared__ float ssc[cCF], ssh[cCF];
  if (threadIdx.x < cCF) {
    float s = bng[threadIdx.x] * rsqrtf(bnv[threadIdx.x] + cEPS);
    ssc[threadIdx.x] = s;
    ssh[threadIdx.x] = bnb[threadIdx.x] - bnm[threadIdx.x] * s;
  }
  __syncthreads();
  int wid = threadIdx.x >> 5, lane = threadIdx.x & 31;
  int tile = blockIdx.x * 4 + wid;
  int bn = tile / (cK / 16);
  int kbase = (tile % (cK / 16)) * 16;
  int row = lane & 15, ko8 = (lane >> 4) << 3;
  int kidx = kbase + row;
  v8f acc[8];
#pragma unroll
  for (int ot = 0; ot < 8; ++ot) acc[ot] = v8f_zero();
#pragma unroll
  for (int kc = 0; kc < 4; ++kc) {
    v16bf a;
#pragma unroll
    for (int e = 0; e < 16; ++e) {
      int c = kc * 32 + e + (ko8 << 1);
      float vfe = feat[(size_t)(bn * cCF + c) * cK + kidx];
      a[e] = (__bf16)fmaxf(vfe * ssc[c] + ssh[c], 0.f);
    }
#pragma unroll
    for (int ot = 0; ot < 8; ++ot) {
      v16bf b = glb_b_frag_n(Wt, cCF, kc * 32, ot * 16);  // W is (o,c) row-major
      acc[ot] = wmma_bf16(a, b, acc[ot]);
    }
  }
  int col = lane & 15, rb = (lane >> 4) << 3;
#pragma unroll
  for (int ot = 0; ot < 8; ++ot)
#pragma unroll
    for (int r = 0; r < 8; ++r) {
      size_t idx = (size_t)(bn * cK + kbase + r + rb) * cD + ot * 16 + col;
      if (accum) outp[idx] += acc[ot][r];
      else outp[idx] = acc[ot][r];
    }
}

// --------------------------------------------------- LN -> linear (WMMA) ------
__global__ void k_ln_gemm(const float* __restrict__ X, const float* __restrict__ g,
                          const float* __restrict__ bvec, const float* __restrict__ Wt,
                          const float* __restrict__ bias, float* __restrict__ outp) {
  __shared__ __bf16 As[4][16 * cD];
  int wid = threadIdx.x >> 5, lane = threadIdx.x & 31;
  int rowbase = (blockIdx.x * 4 + wid) * 16;
#pragma unroll 1
  for (int r = 0; r < 16; ++r) {
    const float* xr = X + (size_t)(rowbase + r) * cD;
    float4 xv = ((const float4*)xr)[lane];
    float s = xv.x + xv.y + xv.z + xv.w;
    float s2 = xv.x * xv.x + xv.y * xv.y + xv.z * xv.z + xv.w * xv.w;
    s = red_add32(s);
    s2 = red_add32(s2);
    float mu = s * (1.f / 128.f);
    float rs = rsqrtf(s2 * (1.f / 128.f) - mu * mu + cEPS);
    int c0 = lane * 4;
    As[wid][r * cD + c0 + 0] = (__bf16)((xv.x - mu) * rs * g[c0 + 0] + bvec[c0 + 0]);
    As[wid][r * cD + c0 + 1] = (__bf16)((xv.y - mu) * rs * g[c0 + 1] + bvec[c0 + 1]);
    As[wid][r * cD + c0 + 2] = (__bf16)((xv.z - mu) * rs * g[c0 + 2] + bvec[c0 + 2]);
    As[wid][r * cD + c0 + 3] = (__bf16)((xv.w - mu) * rs * g[c0 + 3] + bvec[c0 + 3]);
  }
  __syncthreads();
  int col = lane & 15, rb = (lane >> 4) << 3;
#pragma unroll
  for (int ot = 0; ot < 8; ++ot) {
    v8f acc = v8f_zero();
#pragma unroll
    for (int kc = 0; kc < 4; ++kc) {
      v16bf a = lds_a_frag(&As[wid][0], cD, kc * 32);
      v16bf bw = glb_b_frag_k(Wt, cD, kc * 32, ot * 16);  // W is (in,out)
      acc = wmma_bf16(a, bw, acc);
    }
    float bv = bias[ot * 16 + col];
#pragma unroll
    for (int r = 0; r < 8; ++r)
      outp[(size_t)(rowbase + r + rb) * cD + ot * 16 + col] = acc[r] + bv;
  }
}

// ----------------------------------------- flash attention over n*K (WMMA) ----
__global__ void k_attn(const float* __restrict__ qp, const float* __restrict__ kp,
                       const float* __restrict__ vp, float* __restrict__ aout) {
  __shared__ __bf16 Ps[4][16 * 32];
  int wid = threadIdx.x >> 5, lane = threadIdx.x & 31;
  int t = blockIdx.x * 4 + wid;      // (b, m, qtile)
  int qt = t & 63, m = (t >> 6) & 3, b = t >> 8;
  int qbase = qt * 16;
  int col = lane & 15, rb = (lane >> 4) << 3;
  int ko8 = (lane >> 4) << 3, ko16 = (lane >> 4) << 4;
  v8f o0 = v8f_zero(), o1 = v8f_zero();
  float mrow[8], lrow[8];
#pragma unroll
  for (int r = 0; r < 8; ++r) { mrow[r] = -1e30f; lrow[r] = 0.f; }
  const float scale = 0.17677669529663687f;  // 32^-0.5
  for (int n = 0; n < cN; ++n) {
    int bn = b * cN + n;
    v16bf aq;  // per-camera query tile (16 x dh=32)
    {
      const float* qr = qp + (size_t)(bn * cQ + qbase + (lane & 15)) * cD + m * cDH;
#pragma unroll
      for (int e = 0; e < 16; ++e)
        aq[e] = (__bf16)qr[e + ko16];
    }
    for (int kt = 0; kt < 53; ++kt) {   // ceil(1680/32) tiles of 32 keys
      int kb = kt * 32;
      if (kt + 1 < 53) {
        int nk = kb + 32 + col;
        if (nk >= cK) nk = cK - 1;
        __builtin_prefetch((const void*)(kp + (size_t)(bn * cK + nk) * cD + m * cDH + ko16),
                           0, 1);
      }
      int key0 = kb + col, key1 = kb + 16 + col;
      bool val0 = key0 < cK, val1 = key1 < cK;
      int key0c = val0 ? key0 : cK - 1, key1c = val1 ? key1 : cK - 1;
      v16bf bk0, bk1;
      {
        const float* p0 = kp + (size_t)(bn * cK + key0c) * cD + m * cDH + ko16;
        const float* p1 = kp + (size_t)(bn * cK + key1c) * cD + m * cDH + ko16;
#pragma unroll
        for (int e = 0; e < 16; ++e) {
          bk0[e] = (__bf16)(val0 ? p0[e] : 0.f);
          bk1[e] = (__bf16)(val1 ? p1[e] : 0.f);
        }
      }
      v8f s0 = wmma_bf16(aq, bk0, v8f_zero());
      v8f s1 = wmma_bf16(aq, bk1, v8f_zero());
#pragma unroll
      for (int r = 0; r < 8; ++r) {
        float sv0 = val0 ? s0[r] * scale : -1e30f;
        float sv1 = val1 ? s1[r] * scale : -1e30f;
        float mx = red_max16(fmaxf(sv0, sv1));
        float mn = fmaxf(mrow[r], mx);
        float al = __expf(mrow[r] - mn);
        mrow[r] = mn;
        float p0 = __expf(sv0 - mn), p1 = __expf(sv1 - mn);
        float ps = red_add16(p0 + p1);
        lrow[r] = lrow[r] * al + ps;
        o0[r] *= al;
        o1[r] *= al;
        s0[r] = p0;
        s1[r] = p1;
      }
      // C-layout -> A-layout re-distribution for P through LDS
#pragma unroll
      for (int r = 0; r < 8; ++r) {
        Ps[wid][(r + rb) * 32 + col] = (__bf16)s0[r];
        Ps[wid][(r + rb) * 32 + 16 + col] = (__bf16)s1[r];
      }
      __syncthreads();
      v16bf ap = lds_a_frag(&Ps[wid][0], 32, 0);
      v16bf bv0, bv1;
#pragma unroll
      for (int e = 0; e < 16; ++e) {
        int key = kb + ko16 + e;
        bool vld = key < cK;
        int keyc = vld ? key : cK - 1;
        const float* vr = vp + (size_t)(bn * cK + keyc) * cD + m * cDH;
        bv0[e] = (__bf16)(vld ? vr[col] : 0.f);
        bv1[e] = (__bf16)(vld ? vr[16 + col] : 0.f);
      }
      o0 = wmma_bf16(ap, bv0, o0);
      o1 = wmma_bf16(ap, bv1, o1);
      __syncthreads();
    }
  }
#pragma unroll
  for (int r = 0; r < 8; ++r) {
    float inv = 1.f / lrow[r];
    size_t base = (size_t)(b * cQ + qbase + r + rb) * cD + m * cDH;
    aout[base + col] = o0[r] * inv;
    aout[base + 16 + col] = o1[r] * inv;
  }
}

// ------------------------------- GEMM + bias + residual + LayerNorm (WMMA) ----
template <int KD>
__global__ void k_gemm_res_ln(const float* __restrict__ A, const float* __restrict__ Wt,
                              const float* __restrict__ bias, const float* __restrict__ res,
                              int res_t, const float* __restrict__ g,
                              const float* __restrict__ bb, float* __restrict__ outp,
                              int out_t) {
  __shared__ __bf16 As[4][16 * KD];
  int wid = threadIdx.x >> 5, lane = threadIdx.x & 31;
  int rowbase = (blockIdx.x * 4 + wid) * 16;
  for (int idx = lane; idx < 16 * KD; idx += 32)
    As[wid][idx] = (__bf16)A[(size_t)(rowbase + idx / KD) * KD + (idx % KD)];
  __syncthreads();
  v8f acc[8];
#pragma unroll
  for (int ot = 0; ot < 8; ++ot) {
    acc[ot] = v8f_zero();
#pragma unroll
    for (int kc = 0; kc < KD / 32; ++kc)
      acc[ot] = wmma_bf16(lds_a_frag(&As[wid][0], KD, kc * 32),
                          glb_b_frag_k(Wt, cD, kc * 32, ot * 16), acc[ot]);
  }
  int col = lane & 15, rb = (lane >> 4) << 3;
#pragma unroll
  for (int ot = 0; ot < 8; ++ot)
#pragma unroll
    for (int r = 0; r < 8; ++r) {
      int rowg = rowbase + r + rb;
      int o = ot * 16 + col;
      float rv = res_t ? res[(size_t)((rowg >> 10) * cD + o) * cQ + (rowg & (cQ - 1))]
                       : res[(size_t)rowg * cD + o];
      acc[ot][r] += bias[o] + rv;
    }
#pragma unroll
  for (int r = 0; r < 8; ++r) {
    float s = 0.f, s2 = 0.f;
#pragma unroll
    for (int ot = 0; ot < 8; ++ot) {
      float vv = acc[ot][r];
      s += vv;
      s2 += vv * vv;
    }
    s = red_add16(s);
    s2 = red_add16(s2);
    float mu = s * (1.f / 128.f);
    float rs = rsqrtf(s2 * (1.f / 128.f) - mu * mu + cEPS);
    int rowg = rowbase + r + rb;
#pragma unroll
    for (int ot = 0; ot < 8; ++ot) {
      int o = ot * 16 + col;
      float y = (acc[ot][r] - mu) * rs * g[o] + bb[o];
      if (out_t) outp[(size_t)((rowg >> 10) * cD + o) * cQ + (rowg & (cQ - 1))] = y;
      else outp[(size_t)rowg * cD + o] = y;
    }
  }
}

// ------------------------------------------------ GEMM + bias + GELU (WMMA) ---
__global__ void k_gemm_gelu(const float* __restrict__ A, const float* __restrict__ Wt,
                            const float* __restrict__ bias, float* __restrict__ outp) {
  __shared__ __bf16 As[4][16 * cD];
  int wid = threadIdx.x >> 5, lane = threadIdx.x & 31;
  int rowbase = (blockIdx.x * 4 + wid) * 16;
  for (int idx = lane; idx < 16 * cD; idx += 32)
    As[wid][idx] = (__bf16)A[(size_t)(rowbase + idx / cD) * cD + (idx & (cD - 1))];
  __syncthreads();
  int col = lane & 15, rb = (lane >> 4) << 3;
#pragma unroll
  for (int ot = 0; ot < 16; ++ot) {
    v8f acc = v8f_zero();
#pragma unroll
    for (int kc = 0; kc < 4; ++kc)
      acc = wmma_bf16(lds_a_frag(&As[wid][0], cD, kc * 32),
                      glb_b_frag_k(Wt, 2 * cD, kc * 32, ot * 16), acc);
    float bv = bias[ot * 16 + col];
#pragma unroll
    for (int r = 0; r < 8; ++r) {
      float xv = acc[r] + bv;
      float yv = 0.5f * xv * (1.f + erff(xv * 0.70710678118654752f));  // exact GELU
      outp[(size_t)(rowbase + r + rb) * (2 * cD) + ot * 16 + col] = yv;
    }
  }
}

// ------------------------------------------------------------------------------
extern "C" void kernel_launch(void* const* d_in, const int* in_sizes, int n_in,
                              void* d_out, int out_size, void* d_ws, size_t ws_size,
                              hipStream_t stream) {
  (void)in_sizes; (void)n_in; (void)out_size; (void)ws_size;
  const float* x      = (const float*)d_in[0];
  const float* feat   = (const float*)d_in[1];
  const float* I_inv  = (const float*)d_in[2];
  const float* E_inv  = (const float*)d_in[3];
  const float* grid   = (const float*)d_in[4];
  const float* plane  = (const float*)d_in[5];
  const float* fl_g = (const float*)d_in[6],  *fl_b = (const float*)d_in[7];
  const float* fl_m = (const float*)d_in[8],  *fl_v = (const float*)d_in[9];
  const float* fl_w = (const float*)d_in[10];
  const float* fp_g = (const float*)d_in[11], *fp_b = (const float*)d_in[12];
  const float* fp_m = (const float*)d_in[13], *fp_v = (const float*)d_in[14];
  const float* fp_w = (const float*)d_in[15];
  const float* bev_w = (const float*)d_in[16], *bev_b = (const float*)d_in[17];
  const float* img_w = (const float*)d_in[18], *cam_w = (const float*)d_in[19];
  const float* q_lng = (const float*)d_in[20], *q_lnb = (const float*)d_in[21];
  const float* q_w = (const float*)d_in[22], *q_bb = (const float*)d_in[23];
  const float* k_lng = (const float*)d_in[24], *k_lnb = (const float*)d_in[25];
  const float* k_w = (const float*)d_in[26], *k_bb = (const float*)d_in[27];
  const float* v_lng = (const float*)d_in[28], *v_lnb = (const float*)d_in[29];
  const float* v_w = (const float*)d_in[30], *v_bb = (const float*)d_in[31];
  const float* proj_w = (const float*)d_in[32], *proj_b = (const float*)d_in[33];
  const float* pre_g = (const float*)d_in[34], *pre_b = (const float*)d_in[35];
  const float* mlp_w1 = (const float*)d_in[36], *mlp_b1 = (const float*)d_in[37];
  const float* mlp_w2 = (const float*)d_in[38], *mlp_b2 = (const float*)d_in[39];
  const float* post_g = (const float*)d_in[40], *post_b = (const float*)d_in[41];
  float* out = (float*)d_out;
  float* w = (float*)d_ws;

  // workspace layout (floats); ~54 MB total
  const size_t off_c     = 0;
  const size_t off_query = off_c + (size_t)cBN * cD;
  const size_t off_key   = off_query + (size_t)cBN * cQ * cD;
  const size_t off_val   = off_key + (size_t)cBN * cK * cD;
  const size_t off_qp    = off_val + (size_t)cBN * cK * cD;
  const size_t off_kp    = off_qp + (size_t)cBN * cQ * cD;
  const size_t off_vp    = off_kp + (size_t)cBN * cK * cD;
  float* cem   = w + off_c;
  float* query = w + off_query;
  float* keyb  = w + off_key;
  float* valb  = w + off_val;
  float* qp    = w + off_qp;
  float* kp    = w + off_kp;
  float* vp    = w + off_vp;
  // post-attention buffers alias the dead key/val region
  float* attn = keyb;                               // 2*1024*128
  float* z    = keyb + (size_t)cB * cQ * cD;        // 2*1024*128
  float* hbuf = z + (size_t)cB * cQ * cD;           // 2*1024*256

  k_cam_embed<<<cBN, 128, 0, stream>>>(cam_w, E_inv, cem);
  k_bev_query<<<cBN * cQ, 128, 0, stream>>>(bev_w, bev_b, grid, cem, x, query);
  k_img_embed<<<cBN * cK, 128, 0, stream>>>(img_w, I_inv, E_inv, plane, cem, keyb);
  // key = img_embed + conv_fp(feature); val = conv_fl(feature)
  k_conv_wmma<<<(cBN * cK / 16) / 4, 128, 0, stream>>>(feat, fp_g, fp_b, fp_m, fp_v,
                                                       fp_w, keyb, 1);
  k_conv_wmma<<<(cBN * cK / 16) / 4, 128, 0, stream>>>(feat, fl_g, fl_b, fl_m, fl_v,
                                                       fl_w, valb, 0);
  k_ln_gemm<<<(cBN * cQ / 16) / 4, 128, 0, stream>>>(query, q_lng, q_lnb, q_w, q_bb, qp);
  k_ln_gemm<<<(cBN * cK / 16) / 4, 128, 0, stream>>>(keyb, k_lng, k_lnb, k_w, k_bb, kp);
  k_ln_gemm<<<(cBN * cK / 16) / 4, 128, 0, stream>>>(valb, v_lng, v_lnb, v_w, v_bb, vp);
  k_attn<<<(cB * cMH * (cQ / 16)) / 4, 128, 0, stream>>>(qp, kp, vp, attn);
  k_gemm_res_ln<128><<<(cB * cQ / 16) / 4, 128, 0, stream>>>(attn, proj_w, proj_b, x, 1,
                                                             pre_g, pre_b, z, 0);
  k_gemm_gelu<<<(cB * cQ / 16) / 4, 128, 0, stream>>>(z, mlp_w1, mlp_b1, hbuf);
  k_gemm_res_ln<256><<<(cB * cQ / 16) / 4, 128, 0, stream>>>(hbuf, mlp_w2, mlp_b2, z, 0,
                                                             post_g, post_b, out, 1);
}
